// DetrEncoderToSkip_53102975648044
// MI455X (gfx1250) — matrix operand order repaired and checked by reading, not compiled, not run on hardware
//
#include <hip/hip_runtime.h>
#include <hip/hip_bf16.h>
#include <stdint.h>

#define NPTS   16384
#define NB     8
#define NPOINT 128
#define NSAMP  32
#define CIN    256
#define CCAT   259
#define KPAD1  288   // 259 padded to multiple of 32
#define KPAD2  256
#define ROWS   (NB * NPOINT * NSAMP)   // 32768
#define EPSB   1e-5f
#define R2     0.16f

#define PANEL1_BYTES ((KPAD1 / 32) * 16 * 32 * 32)   // 147456
#define PANEL2_BYTES ((KPAD2 / 32) * 16 * 32 * 32)   // 131072

typedef __attribute__((ext_vector_type(16))) __bf16 bf16x16;
typedef __attribute__((ext_vector_type(8)))  float  f32x8;
typedef int v4i __attribute__((ext_vector_type(4)));
typedef __attribute__((address_space(1))) v4i gbl_v4i;   // prints as __device__
typedef __attribute__((address_space(3))) v4i lds_v4i;   // prints as __shared__

#if defined(__has_builtin)
#if __has_builtin(__builtin_amdgcn_global_load_async_to_lds_b128)
#define USE_ASYNC_LDS 1
#endif
#if __has_builtin(__builtin_amdgcn_s_wait_asynccnt)
#define HAVE_WAIT_ASYNC_BUILTIN 1
#endif
#endif

union Frag {
  bf16x16 bf;
  uint32_t u[8];
  uint4 q[2];
};

static __device__ __forceinline__ unsigned short f2bf(float f) {
  union { float f; uint32_t u; } v; v.f = f;
  uint32_t u = v.u;
  u += 0x7FFFu + ((u >> 16) & 1u);   // round-to-nearest-even
  return (unsigned short)(u >> 16);
}

static __device__ __forceinline__ int xsh_i(int v, int m) {
  int l = ((int)(threadIdx.x & 31)) ^ m;
  return __builtin_amdgcn_ds_bpermute(l << 2, v);
}
static __device__ __forceinline__ float xsh_f(float v, int m) {
  return __int_as_float(xsh_i(__float_as_int(v), m));
}

// Stage `nbytes` (multiple of 16*blockDim) of global panel into LDS offset 0.
// Uses CDNA5 async global->LDS DMA when the builtin exists (ASYNCcnt path),
// else a plain load+ds_store copy.  Kernels using this must have NO static
// __shared__ so the dynamic LDS base offset is 0.
static __device__ __forceinline__ void stage_panel(const unsigned short* panel,
                                                   char* smem_generic,
                                                   int nbytes) {
#if defined(USE_ASYNC_LDS)
  for (int off = threadIdx.x * 16; off < nbytes; off += blockDim.x * 16) {
    gbl_v4i* gsrc = (gbl_v4i*)(uintptr_t)((uintptr_t)panel + off);
    lds_v4i* ldst = (lds_v4i*)(uintptr_t)off;   // dynamic LDS base == 0
    __builtin_amdgcn_global_load_async_to_lds_b128(gsrc, ldst, 0, 0);
  }
#if defined(HAVE_WAIT_ASYNC_BUILTIN)
  __builtin_amdgcn_s_wait_asynccnt(0);
#else
  asm volatile("s_wait_asynccnt 0x0" ::: "memory");
#endif
#else
  for (int off = threadIdx.x * 16; off < nbytes; off += blockDim.x * 16) {
    *(uint4*)(smem_generic + off) =
        *(const uint4*)((const char*)panel + off);
  }
#endif
  __syncthreads();
}

// ---------------------------------------------------------------------------
// Kernel 1: farthest point sampling. One workgroup per batch, dist in VGPRs.
// ---------------------------------------------------------------------------
__global__ void fps_kernel(const float* __restrict__ xyz,
                           int* __restrict__ inds,
                           float* __restrict__ new_xyz) {
  __shared__ float redv[16];
  __shared__ int   redi[16];
  __shared__ int   s_far;
  const int b   = blockIdx.x;
  const int tid = threadIdx.x;            // 512 threads
  const int lane = tid & 31, wave = tid >> 5;
  const float* xb = xyz + (size_t)b * NPTS * 3;

  float dloc[32];
#pragma unroll
  for (int j = 0; j < 32; ++j) dloc[j] = 1e10f;

  if (tid == 0) s_far = 0;
  __syncthreads();

  for (int it = 0; it < NPOINT; ++it) {
    const int far = s_far;
    if (tid == 0) {
      inds[b * NPOINT + it] = far;
      new_xyz[(b * NPOINT + it) * 3 + 0] = xb[far * 3 + 0];
      new_xyz[(b * NPOINT + it) * 3 + 1] = xb[far * 3 + 1];
      new_xyz[(b * NPOINT + it) * 3 + 2] = xb[far * 3 + 2];
    }
    const float cx = xb[far * 3 + 0], cy = xb[far * 3 + 1], cz = xb[far * 3 + 2];
    float best = -1.0f; int bidx = 0x7fffffff;
#pragma unroll 4
    for (int j = 0; j < 32; ++j) {
      const int i = tid + j * 512;
      const float dx = xb[i * 3 + 0] - cx;
      const float dy = xb[i * 3 + 1] - cy;
      const float dz = xb[i * 3 + 2] - cz;
      const float d  = dx * dx + dy * dy + dz * dz;
      const float nd = fminf(dloc[j], d);
      dloc[j] = nd;
      if (nd > best) { best = nd; bidx = i; }
    }
    // wave32 argmax (prefer smaller index on ties)
#pragma unroll
    for (int off = 16; off; off >>= 1) {
      const float ov = xsh_f(best, off);
      const int   oi = xsh_i(bidx, off);
      if (ov > best || (ov == best && oi < bidx)) { best = ov; bidx = oi; }
    }
    if (lane == 0) { redv[wave] = best; redi[wave] = bidx; }
    __syncthreads();
    if (tid < 32) {
      float bv = (tid < 16) ? redv[tid] : -1.0f;
      int   bi = (tid < 16) ? redi[tid] : 0x7fffffff;
#pragma unroll
      for (int off = 16; off; off >>= 1) {
        const float ov = xsh_f(bv, off);
        const int   oi = xsh_i(bi, off);
        if (ov > bv || (ov == bv && oi < bi)) { bv = ov; bi = oi; }
      }
      if (tid == 0) s_far = bi;
    }
    __syncthreads();
  }
}

// ---------------------------------------------------------------------------
// Kernel 2: ball query. One wave per centroid; ordered append via ballot.
// ---------------------------------------------------------------------------
__global__ void ballquery_kernel(const float* __restrict__ xyz,
                                 const float* __restrict__ new_xyz,
                                 int* __restrict__ idx) {
  const int gw   = (int)((blockIdx.x * blockDim.x + threadIdx.x) >> 5);
  const int lane = threadIdx.x & 31;
  if (gw >= NB * NPOINT) return;
  const int b = gw / NPOINT;
  const float* xb = xyz + (size_t)b * NPTS * 3;
  const float cx = new_xyz[gw * 3 + 0];
  const float cy = new_xyz[gw * 3 + 1];
  const float cz = new_xyz[gw * 3 + 2];
  int* out = idx + (size_t)gw * NSAMP;

  int cnt = 0;
  int firstIdx = NPTS - 1;
  for (int base = 0; base < NPTS && cnt < NSAMP; base += 32) {
    const int i = base + lane;
    const float dx = xb[i * 3 + 0] - cx;
    const float dy = xb[i * 3 + 1] - cy;
    const float dz = xb[i * 3 + 2] - cz;
    const bool within = (dx * dx + dy * dy + dz * dz) < R2;
    const unsigned m32 = __builtin_amdgcn_ballot_w32(within);
    if (within) {
      const int pos = cnt + __popc(m32 & ((1u << lane) - 1u));
      if (pos < NSAMP) out[pos] = i;
    }
    if (cnt == 0 && m32) firstIdx = base + __ffs(m32) - 1;
    cnt += __popc(m32);
  }
  cnt = min(cnt, NSAMP);
  if (lane >= cnt) out[lane] = firstIdx;   // pad with first hit
}

// ---------------------------------------------------------------------------
// Kernel 3: gather + concat -> bf16 activation rows padded to KPAD1.
// One wave per row (b,s,k).  Row = ((b*NPOINT+s)*NSAMP + k).
// ---------------------------------------------------------------------------
__global__ void buildx_kernel(const float* __restrict__ xyz,
                              const float* __restrict__ feats,
                              const float* __restrict__ new_xyz,
                              const int* __restrict__ idx,
                              unsigned short* __restrict__ xpad) {
  const int row  = (int)((blockIdx.x * blockDim.x + threadIdx.x) >> 5);
  const int lane = threadIdx.x & 31;
  if (row >= ROWS) return;
  const int g  = row >> 5;          // group = b*NPOINT + s
  const int b  = g / NPOINT;
  const int pt = idx[row];
  const float* p = xyz + ((size_t)b * NPTS + pt) * 3;
  unsigned short* orow = xpad + (size_t)row * KPAD1;
  if (lane == 0) {
    orow[0] = f2bf(p[0] - new_xyz[g * 3 + 0]);
    orow[1] = f2bf(p[1] - new_xyz[g * 3 + 1]);
    orow[2] = f2bf(p[2] - new_xyz[g * 3 + 2]);
  }
  const float* f = feats + ((size_t)b * NPTS + pt) * CIN;
#pragma unroll
  for (int c = lane; c < CIN; c += 32) orow[3 + c] = f2bf(f[c]);
  if (lane < KPAD1 - CCAT) orow[CCAT + lane] = 0;  // zero K padding
}

// ---------------------------------------------------------------------------
// Kernel 4: pack weights [O, Cin] -> fragment-ordered bf16 B-panel.
// panel dword t = ((kb*16+nb)*32 + L)*8 + v holds pair (W[n,k], W[n,k+1]),
// n = nb*16 + (L&15), k = kb*32 + (L>>4)*8 + (v>>2)*16 + 2(v&3).   (ISA 16-bit B layout)
// ---------------------------------------------------------------------------
__global__ void packw_kernel(const float* __restrict__ w, int cin, int kpad,
                             uint32_t* __restrict__ panel) {
  const int t = blockIdx.x * blockDim.x + threadIdx.x;
  const int total = (kpad / 32) * 16 * 32 * 8;
  if (t >= total) return;
  const int v  = t & 7;
  const int L  = (t >> 3) & 31;
  const int nb = (t >> 8) & 15;
  const int kb = t >> 12;
  const int n  = nb * 16 + (L & 15);
  const int k  = kb * 32 + (L >> 4) * 8 + (v >> 2) * 16 + 2 * (v & 3);
  const float a0 = (k     < cin) ? w[(size_t)n * cin + k    ] : 0.0f;
  const float a1 = (k + 1 < cin) ? w[(size_t)n * cin + k + 1] : 0.0f;
  panel[t] = (uint32_t)f2bf(a0) | ((uint32_t)f2bf(a1) << 16);
}

// ---------------------------------------------------------------------------
// Fragment loads.  ISA 16-bit A layout: lane L -> m = L&15;
// VGPR0-3 = K (L<16 ? 0..7 : 8..15), VGPR4-7 = +16.
// ---------------------------------------------------------------------------
static __device__ __forceinline__ Frag load_a(const unsigned short* rowbase,
                                              int stride, int kb, int lane) {
  Frag a;
  const int c0 = kb * 32 + ((lane >> 4) << 3);
  const unsigned short* p = rowbase + (size_t)(lane & 15) * stride + c0;
  a.q[0] = *(const uint4*)(p);
  a.q[1] = *(const uint4*)(p + 16);
  return a;
}
// B fragment from LDS-staged fragment-ordered panel (two ds_load_b128).
static __device__ __forceinline__ Frag load_b_lds(const char* smem,
                                                  int kb, int nb, int lane) {
  Frag bfr;
  const char* p = smem + ((size_t)((kb * 16 + nb) * 32 + lane)) * 32;
  bfr.q[0] = *(const uint4*)(p);
  bfr.q[1] = *(const uint4*)(p + 16);
  return bfr;
}

// ---------------------------------------------------------------------------
// Kernel 5: GEMM1 + BN1 + ReLU -> h1 bf16 [ROWS, 256].
// Panel staged in LDS via async copies.  One wave computes a 32x64 super-tile
// (8 accumulators); per K-step: 4 global b128 (A) + 8 ds b128 (B) -> 8 WMMA.
// Block = 8 waves: w>>2 selects 32-row half, w&3 selects 64-col strip.
// ---------------------------------------------------------------------------
__global__ void gemm1_kernel(const unsigned short* __restrict__ xpad,
                             const unsigned short* __restrict__ panel1,
                             const float* __restrict__ gamma,
                             const float* __restrict__ beta,
                             const float* __restrict__ mean,
                             const float* __restrict__ var,
                             unsigned short* __restrict__ h1) {
  extern __shared__ char smem[];
  stage_panel(panel1, smem, PANEL1_BYTES);

  const int w    = threadIdx.x >> 5;
  const int lane = threadIdx.x & 31;
  const int cs   = w & 3;
  const int r0   = blockIdx.x * 64 + (w >> 2) * 32;   // first of 32 rows

  f32x8 acc[2][4];
#pragma unroll
  for (int r = 0; r < 2; ++r)
#pragma unroll
    for (int i = 0; i < 4; ++i)
      acc[r][i] = (f32x8){0.f, 0.f, 0.f, 0.f, 0.f, 0.f, 0.f, 0.f};

  const unsigned short* arow0 = xpad + (size_t)r0 * KPAD1;
  const unsigned short* arow1 = arow0 + (size_t)16 * KPAD1;
#pragma unroll
  for (int kb = 0; kb < KPAD1 / 32; ++kb) {
    const Frag A0 = load_a(arow0, KPAD1, kb, lane);
    const Frag A1 = load_a(arow1, KPAD1, kb, lane);
#pragma unroll
    for (int nb = 0; nb < 4; ++nb) {
      const Frag Bf = load_b_lds(smem, kb, cs * 4 + nb, lane);
      acc[0][nb] = __builtin_amdgcn_wmma_f32_16x16x32_bf16(
          false, A0.bf, false, Bf.bf, (short)0, acc[0][nb], false, false);
      acc[1][nb] = __builtin_amdgcn_wmma_f32_16x16x32_bf16(
          false, A1.bf, false, Bf.bf, (short)0, acc[1][nb], false, false);
    }
  }
  // epilogue: BN + ReLU, store bf16.  C layout: n = lane&15, m = v + (lane>>4)*8
#pragma unroll
  for (int nb = 0; nb < 4; ++nb) {
    const int n  = (cs * 4 + nb) * 16 + (lane & 15);
    const float sc = gamma[n] * rsqrtf(var[n] + EPSB);
    const float sh = beta[n] - mean[n] * sc;
#pragma unroll
    for (int r = 0; r < 2; ++r)
#pragma unroll
      for (int v = 0; v < 8; ++v) {
        const int m = r0 + r * 16 + v + ((lane >> 4) << 3);
        const float val = fmaxf(acc[r][nb][v] * sc + sh, 0.0f);
        h1[(size_t)m * 256 + n] = f2bf(val);
      }
  }
}

// ---------------------------------------------------------------------------
// Kernel 6: GEMM2 + BN2 + ReLU + max-over-32-samples, writes [B,256,128].
// A wave's 32-row super-tile is exactly one centroid group, so the K=32 max
// reduces fully in-wave (v-loop + lane^16 bpermute).  Block = 8 waves =
// 2 groups x 4 col strips; panel2 staged in LDS.
// ---------------------------------------------------------------------------
__global__ void gemm2_kernel(const unsigned short* __restrict__ h1,
                             const unsigned short* __restrict__ panel2,
                             const float* __restrict__ gamma,
                             const float* __restrict__ beta,
                             const float* __restrict__ mean,
                             const float* __restrict__ var,
                             float* __restrict__ out) {
  extern __shared__ char smem[];
  stage_panel(panel2, smem, PANEL2_BYTES);

  const int w    = threadIdx.x >> 5;
  const int lane = threadIdx.x & 31;
  const int cs   = w & 3;
  const int g    = blockIdx.x * 2 + (w >> 2);   // centroid group

  f32x8 acc[2][4];
#pragma unroll
  for (int r = 0; r < 2; ++r)
#pragma unroll
    for (int i = 0; i < 4; ++i)
      acc[r][i] = (f32x8){0.f, 0.f, 0.f, 0.f, 0.f, 0.f, 0.f, 0.f};

  const unsigned short* arow0 = h1 + (size_t)g * 32 * KPAD2;
  const unsigned short* arow1 = arow0 + (size_t)16 * KPAD2;
#pragma unroll
  for (int kb = 0; kb < KPAD2 / 32; ++kb) {
    const Frag A0 = load_a(arow0, KPAD2, kb, lane);
    const Frag A1 = load_a(arow1, KPAD2, kb, lane);
#pragma unroll
    for (int nb = 0; nb < 4; ++nb) {
      const Frag Bf = load_b_lds(smem, kb, cs * 4 + nb, lane);
      acc[0][nb] = __builtin_amdgcn_wmma_f32_16x16x32_bf16(
          false, A0.bf, false, Bf.bf, (short)0, acc[0][nb], false, false);
      acc[1][nb] = __builtin_amdgcn_wmma_f32_16x16x32_bf16(
          false, A1.bf, false, Bf.bf, (short)0, acc[1][nb], false, false);
    }
  }

  const int b = g >> 7;            // g / NPOINT
  const int s = g & 127;           // g % NPOINT
#pragma unroll
  for (int nb = 0; nb < 4; ++nb) {
    const int n  = (cs * 4 + nb) * 16 + (lane & 15);
    const float sc = gamma[n] * rsqrtf(var[n] + EPSB);
    const float sh = beta[n] - mean[n] * sc;
    float mx = 0.0f;   // ReLU outputs are >= 0
#pragma unroll
    for (int r = 0; r < 2; ++r)
#pragma unroll
      for (int v = 0; v < 8; ++v)
        mx = fmaxf(mx, fmaxf(acc[r][nb][v] * sc + sh, 0.0f));
    mx = fmaxf(mx, xsh_f(mx, 16));   // combine both m-halves (lane^16, same n)
    if (lane < 16) {
      const int c = (cs * 4 + nb) * 16 + lane;
      out[((size_t)b * 256 + c) * NPOINT + s] = mx;
    }
  }
}

// ---------------------------------------------------------------------------
extern "C" void kernel_launch(void* const* d_in, const int* in_sizes, int n_in,
                              void* d_out, int out_size, void* d_ws, size_t ws_size,
                              hipStream_t stream) {
  const float* xyz    = (const float*)d_in[0];
  const float* feats  = (const float*)d_in[1];
  const float* w1     = (const float*)d_in[2];
  const float* gamma1 = (const float*)d_in[3];
  const float* beta1  = (const float*)d_in[4];
  const float* mean1  = (const float*)d_in[5];
  const float* var1   = (const float*)d_in[6];
  const float* w2     = (const float*)d_in[7];
  const float* gamma2 = (const float*)d_in[8];
  const float* beta2  = (const float*)d_in[9];
  const float* mean2  = (const float*)d_in[10];
  const float* var2   = (const float*)d_in[11];
  float* out = (float*)d_out;
  char* ws = (char*)d_ws;

  // Workspace layout (16B-aligned offsets)
  int*            inds    = (int*)           (ws + 0);          //   4 KB
  float*          new_xyz = (float*)         (ws + 4096);       //  12 KB
  int*            idx     = (int*)           (ws + 16384);      // 128 KB
  unsigned short* panel1  = (unsigned short*)(ws + 147456);     // 144 KB
  unsigned short* panel2  = (unsigned short*)(ws + 294912);     // 128 KB
  unsigned short* xpad    = (unsigned short*)(ws + 425984);     // ~18.9 MB
  unsigned short* h1      = (unsigned short*)(ws + 19300352);   // ~16.8 MB

  // Weight panels (independent of point pipeline)
  packw_kernel<<<(9 * 4096 + 255) / 256, 256, 0, stream>>>(w1, CCAT, KPAD1,
                                                           (uint32_t*)panel1);
  packw_kernel<<<(8 * 4096 + 255) / 256, 256, 0, stream>>>(w2, 256, KPAD2,
                                                           (uint32_t*)panel2);

  // Point pipeline
  fps_kernel<<<NB, 512, 0, stream>>>(xyz, inds, new_xyz);
  ballquery_kernel<<<(NB * NPOINT * 32 + 255) / 256, 256, 0, stream>>>(
      xyz, new_xyz, idx);
  buildx_kernel<<<(ROWS * 32 + 255) / 256, 256, 0, stream>>>(
      xyz, feats, new_xyz, idx, xpad);

  // WMMA GEMMs (weight panel staged to LDS per workgroup via async DMA)
  gemm1_kernel<<<ROWS / 64, 256, PANEL1_BYTES, stream>>>(
      xpad, panel1, gamma1, beta1, mean1, var1, h1);
  gemm2_kernel<<<NB * NPOINT / 2, 256, PANEL2_BYTES, stream>>>(
      h1, panel2, gamma2, beta2, mean2, var2, out);
}
